// MultiHeadAttention_12034498363602
// MI455X (gfx1250) — matrix operand order repaired and checked by reading, not compiled, not run on hardware
//
#include <hip/hip_runtime.h>
#include <hip/hip_bf16.h>
#include <math.h>

// ---------------------------------------------------------------------------
// MHA forward for MI455X (gfx1250, wave32, WMMA 16x16x32 f16 -> f32 acc).
// Phases:
//   1) proj_kernel    : Q/K/V projections (f32 in -> f16 head layouts), K transposed,
//                       softmax scale folded into Q.
//   2) attn_kernel    : fused scores + softmax + ctx. One WG per (b,h,16-row block).
//                       16x2048 f32 score stripe lives in LDS (128KB < 320KB/WGP),
//                       normalized attn written to HBM exactly once.
//   3) outproj_kernel : concat @ Wo + bo -> f32 out.
// ---------------------------------------------------------------------------

typedef __attribute__((ext_vector_type(16))) _Float16 v16h;
typedef __attribute__((ext_vector_type(8)))  float    v8f;

static constexpr int BB   = 2;
static constexpr int SS   = 2048;
static constexpr int DM   = 1024;
static constexpr int NH   = 16;
static constexpr int DH   = 64;
static constexpr int ROWS = BB * SS;                       // 4096
static constexpr size_t HEAD_ELEMS = (size_t)BB * NH * SS * DH; // 4,194,304 halfs
static constexpr int LDSP = SS + 8;                        // padded LDS pitch (floats)

__device__ __forceinline__ v8f wmma_f16(v16h a, v16h b, v8f c) {
  // D = A(16x32 f16) * B(32x16 f16) + C(16x16 f32)
  return __builtin_amdgcn_wmma_f32_16x16x32_f16(false, a, false, b, (short)0, c,
                                                false, false);
}

// A operand (16x32): lane L -> row m=L&15, group g=L>>4 holds K = g*8+i and 16+g*8+i.
__device__ __forceinline__ v16h load_a_f16(const _Float16* __restrict__ base,
                                           int pitch, int k0) {
  const int lane = threadIdx.x & 31;
  const int m = lane & 15, g = lane >> 4;
  const _Float16* p = base + (size_t)m * pitch + k0 + g * 8;
  v16h a;
#pragma unroll
  for (int i = 0; i < 8; ++i) a[i] = p[i];
#pragma unroll
  for (int i = 0; i < 8; ++i) a[8 + i] = p[16 + i];
  return a;
}

__device__ __forceinline__ v16h load_a_f32(const float* __restrict__ base,
                                           int pitch, int k0) {
  const int lane = threadIdx.x & 31;
  const int m = lane & 15, g = lane >> 4;
  const float* p = base + (size_t)m * pitch + k0 + g * 8;
  v16h a;
#pragma unroll
  for (int i = 0; i < 8; ++i) a[i] = (_Float16)p[i];
#pragma unroll
  for (int i = 0; i < 8; ++i) a[8 + i] = (_Float16)p[16 + i];
  return a;
}

// B operand (32x16): lane L -> col n=L&15, group g=L>>4 holds K = g*8+i and 16+g*8+i.
__device__ __forceinline__ v16h load_b_f16(const _Float16* __restrict__ base,
                                           int pitch, int k0, int n0) {
  const int lane = threadIdx.x & 31;
  const int n = lane & 15, g = lane >> 4;
  const _Float16* p = base + (size_t)(k0 + g * 8) * pitch + n0 + n;
  v16h b;
#pragma unroll
  for (int i = 0; i < 8; ++i) b[i] = p[(size_t)i * pitch];
#pragma unroll
  for (int i = 0; i < 8; ++i) b[8 + i] = p[(size_t)(16 + i) * pitch];
  return b;
}

__device__ __forceinline__ v16h load_b_f32(const float* __restrict__ base,
                                           int pitch, int k0, int n0) {
  const int lane = threadIdx.x & 31;
  const int n = lane & 15, g = lane >> 4;
  const float* p = base + (size_t)(k0 + g * 8) * pitch + n0 + n;
  v16h b;
#pragma unroll
  for (int i = 0; i < 8; ++i) b[i] = (_Float16)p[(size_t)i * pitch];
#pragma unroll
  for (int i = 0; i < 8; ++i) b[8 + i] = (_Float16)p[(size_t)(16 + i) * pitch];
  return b;
}

// ---------------------------------------------------------------------------
// Phase 1: QKV projections.  grid = (4096, 1, 3), block = 128 (4 waves).
// mode 0: Qh[b,h,s,d] (scaled by 1/8), mode 1: KhT[b,h,d,s], mode 2: Vh[b,h,s,d]
// ---------------------------------------------------------------------------
__global__ void proj_kernel(const float* __restrict__ q, const float* __restrict__ k,
                            const float* __restrict__ v,
                            const float* __restrict__ wq, const float* __restrict__ bq,
                            const float* __restrict__ wk, const float* __restrict__ bk,
                            const float* __restrict__ wv, const float* __restrict__ bv,
                            _Float16* __restrict__ qh, _Float16* __restrict__ khT,
                            _Float16* __restrict__ vh) {
  const int wave = threadIdx.x >> 5;
  const int tile = blockIdx.x * 4 + wave;        // 16384 tiles
  const int mt = tile >> 6;                      // 256 row tiles
  const int nt = tile & 63;                      // 64  col tiles
  const int mode = blockIdx.z;

  const float* X  = (mode == 0) ? q  : (mode == 1) ? k  : v;
  const float* W  = (mode == 0) ? wq : (mode == 1) ? wk : wv;
  const float* Bv = (mode == 0) ? bq : (mode == 1) ? bk : bv;

  const float* xrow = X + (size_t)(mt * 16) * DM;
  v8f acc = {};
  for (int kc = 0; kc < DM; kc += 32) {
    v16h a = load_a_f32(xrow, DM, kc);
    v16h b = load_b_f32(W, DM, kc, nt * 16);
    acc = wmma_f16(a, b, acc);
  }

  const int lane = threadIdx.x & 31;
  const int col = lane & 15, rowoff = (lane >> 4) * 8;
  const int ng = nt * 16 + col;
  const float bias = Bv[ng];
  const int h = ng >> 6, d = ng & 63;
#pragma unroll
  for (int r = 0; r < 8; ++r) {
    const int rg = mt * 16 + rowoff + r;
    const int bb = rg >> 11, s = rg & (SS - 1);
    const size_t bh = (size_t)(bb * NH + h);
    const float val = acc[r] + bias;
    if (mode == 0)      qh [(bh * SS + s) * DH + d] = (_Float16)(val * 0.125f);
    else if (mode == 1) khT[(bh * DH + d) * SS + s] = (_Float16)val;
    else                vh [(bh * SS + s) * DH + d] = (_Float16)val;
  }
}

// ---------------------------------------------------------------------------
// Phase 2: fused scores + softmax + ctx.
// grid = 32*(2048/16) = 4096 WGs, block = 128 (4 waves).
// LDS: 16 x LDSP f32 score stripe (131,584 B dynamic LDS; CDNA5 WGP has 320KB).
// ---------------------------------------------------------------------------
__global__ void attn_kernel(const _Float16* __restrict__ qh,
                            const _Float16* __restrict__ khT,
                            const _Float16* __restrict__ vh,
                            float* __restrict__ attn,
                            _Float16* __restrict__ concat) {
  extern __shared__ float smem[];                // [16][LDSP]
  const int wave = threadIdx.x >> 5;
  const int lane = threadIdx.x & 31;
  const int col = lane & 15, rowoff = (lane >> 4) * 8;

  const int bh = blockIdx.x >> 7;                // 0..31
  const int mt = blockIdx.x & 127;               // 16-row block of queries

  const _Float16* qb = qh  + (size_t)bh * SS * DH + (size_t)(mt * 16) * DH;
  const _Float16* kb = khT + (size_t)bh * DH * SS;
  const _Float16* vb = vh  + (size_t)bh * SS * DH;

  // ---- scores: Q(16x64) @ K^T(64x2048) -> LDS -------------------------------
  const v16h a0 = load_a_f16(qb, DH, 0);
  const v16h a1 = load_a_f16(qb, DH, 32);
  for (int nt = wave; nt < SS / 16; nt += 4) {
    v8f acc = {};
    acc = wmma_f16(a0, load_b_f16(kb, SS, 0,  nt * 16), acc);
    acc = wmma_f16(a1, load_b_f16(kb, SS, 32, nt * 16), acc);
#pragma unroll
    for (int r = 0; r < 8; ++r)
      smem[(rowoff + r) * LDSP + nt * 16 + col] = acc[r];
  }
  __syncthreads();

  // ---- softmax over each of the 16 rows (one wave per row, round-robin) -----
  float* attn_base = attn + (size_t)bh * SS * SS + (size_t)(mt * 16) * SS;
  for (int row = wave; row < 16; row += 4) {
    float* rp = smem + row * LDSP;
    float mx = -1e30f;
    for (int j = lane; j < SS; j += 32) mx = fmaxf(mx, rp[j]);
#pragma unroll
    for (int off = 16; off > 0; off >>= 1) mx = fmaxf(mx, __shfl_xor(mx, off, 32));
    float sum = 0.f;
    for (int j = lane; j < SS; j += 32) {
      const float e = __expf(rp[j] - mx);
      rp[j] = e;
      sum += e;
    }
#pragma unroll
    for (int off = 16; off > 0; off >>= 1) sum += __shfl_xor(sum, off, 32);
    const float inv = 1.0f / sum;
    float* gout = attn_base + (size_t)row * SS;
    for (int j = lane; j < SS; j += 32) {
      const float p = rp[j] * inv;
      rp[j] = p;                                 // kept for ctx GEMM
      gout[j] = p;                               // attn written to HBM once
    }
  }
  __syncthreads();

  // ---- ctx: attn(16x2048, from LDS) @ V(2048x64); wave owns one 16-col tile -
  const int n0 = wave * 16;
  v8f acc = {};
  for (int kc = 0; kc < SS; kc += 32) {
    // A operand straight out of LDS (f32 -> f16)
    const int m = lane & 15, g = lane >> 4;
    const float* p = smem + m * LDSP + kc + g * 8;
    v16h a;
#pragma unroll
    for (int i = 0; i < 8; ++i) a[i] = (_Float16)p[i];
#pragma unroll
    for (int i = 0; i < 8; ++i) a[8 + i] = (_Float16)p[16 + i];
    acc = wmma_f16(a, load_b_f16(vb, DH, kc, n0), acc);
  }

  const int b = bh >> 4, h = bh & 15;
#pragma unroll
  for (int r = 0; r < 8; ++r) {
    const int s = mt * 16 + rowoff + r;
    concat[((size_t)(b * SS + s)) * DM + h * DH + n0 + col] = (_Float16)acc[r];
  }
}

// ---------------------------------------------------------------------------
// Phase 3: out = concat @ Wo + bo.  grid = 4096, block = 128 (4 waves).
// ---------------------------------------------------------------------------
__global__ void outproj_kernel(const _Float16* __restrict__ concat,
                               const float* __restrict__ wo,
                               const float* __restrict__ bo,
                               float* __restrict__ out) {
  const int wave = threadIdx.x >> 5;
  const int tile = blockIdx.x * 4 + wave;        // 16384 tiles
  const int mt = tile >> 6;
  const int nt = tile & 63;

  const _Float16* arow = concat + (size_t)(mt * 16) * DM;
  v8f acc = {};
  for (int kc = 0; kc < DM; kc += 32) {
    v16h a = load_a_f16(arow, DM, kc);
    v16h b = load_b_f32(wo, DM, kc, nt * 16);
    acc = wmma_f16(a, b, acc);
  }

  const int lane = threadIdx.x & 31;
  const int col = lane & 15, rowoff = (lane >> 4) * 8;
  const float bias = bo[nt * 16 + col];
#pragma unroll
  for (int r = 0; r < 8; ++r)
    out[(size_t)(mt * 16 + rowoff + r) * DM + nt * 16 + col] = acc[r] + bias;
}

// ---------------------------------------------------------------------------
extern "C" void kernel_launch(void* const* d_in, const int* in_sizes, int n_in,
                              void* d_out, int out_size, void* d_ws, size_t ws_size,
                              hipStream_t stream) {
  (void)in_sizes; (void)n_in; (void)out_size; (void)ws_size;

  const float* q  = (const float*)d_in[0];
  const float* k  = (const float*)d_in[1];
  const float* v  = (const float*)d_in[2];
  // d_in[3] = mask (unused by reference)
  const float* wq = (const float*)d_in[4];
  const float* bq = (const float*)d_in[5];
  const float* wk = (const float*)d_in[6];
  const float* bk = (const float*)d_in[7];
  const float* wv = (const float*)d_in[8];
  const float* bv = (const float*)d_in[9];
  const float* wo = (const float*)d_in[10];
  const float* bo = (const float*)d_in[11];

  float* out  = (float*)d_out;                       // [2,2048,1024]
  float* attn = out + (size_t)ROWS * DM;             // [2,16,2048,2048]

  _Float16* ws     = (_Float16*)d_ws;                // 32 MB total
  _Float16* qh     = ws;                             // [2,16,2048,64]
  _Float16* khT    = ws + HEAD_ELEMS;                // [2,16,64,2048]
  _Float16* vh     = ws + 2 * HEAD_ELEMS;            // [2,16,2048,64]
  _Float16* concat = ws + 3 * HEAD_ELEMS;            // [4096,1024]

  proj_kernel<<<dim3(4096, 1, 3), 128, 0, stream>>>(q, k, v, wq, bq, wk, bk,
                                                    wv, bv, qh, khT, vh);

  const size_t lds_bytes = (size_t)16 * LDSP * sizeof(float);   // 131,584 B
  attn_kernel<<<dim3(4096), 128, lds_bytes, stream>>>(qh, khT, vh, attn, concat);

  outproj_kernel<<<dim3(4096), 128, 0, stream>>>(concat, wo, bo, out);
}
